// GaussianSelfAttention_59863254171747
// MI455X (gfx1250) — matrix operand order repaired and checked by reading, not compiled, not run on hardware
//
#include <hip/hip_runtime.h>
#include <hip/hip_bf16.h>
#include <stdint.h>

// ---------------------------------------------------------------------------
// GaussianSelfAttention for MI455X (gfx1250, wave32, WMMA + async LDS copies).
//   probs  : softmax over Gaussian scores  -> d_out[OUT_ELEMS ..]
//   vals   : probs x hidden  (M=20736,K=2304,N=2048)  -> WMMA bf16 hi/lo split
//   out    : vals  x value_w.T + bias (M=18432,K=2304,N=256) -> WMMA
// B matrices packed N-major ([n][kk]) so async global->LDS copies are b128.
// ---------------------------------------------------------------------------

typedef __attribute__((ext_vector_type(16))) __bf16 v16bf;
typedef __attribute__((ext_vector_type(8)))  float  v8f;

// pointer types matching the async-to-LDS builtin's parameters
// (global = AS1, LDS = AS3, pointee = GCC-vector int4 per hipcc diagnostic)
typedef int v4i_vs __attribute__((vector_size(16)));
typedef __attribute__((address_space(1))) v4i_vs* g_v4i_p;
typedef __attribute__((address_space(3))) v4i_vs* l_v4i_p;

#define NHEADS  9
#define HW      48
#define CH      256
#define BATCH   8
#define KDIM    2304            // 48*48 == 9*256
#define M1      20736           // 48*48*9
#define N1      2048            // 8*256
#define M2      18432           // 8*48*48
#define N2      256
#define OUT_ELEMS 4718592       // 8*256*48*48

#if __has_builtin(__builtin_amdgcn_global_load_async_to_lds_b128)
#define HAVE_ASYNC 1
#else
#define HAVE_ASYNC 0
#endif

// one 16-byte global->LDS copy (async if available)
__device__ __forceinline__ void copy16_to_lds(const __bf16* g, __bf16* l) {
#if HAVE_ASYNC
  __builtin_amdgcn_global_load_async_to_lds_b128(
      (g_v4i_p)(uintptr_t)g, (l_v4i_p)(unsigned)(uintptr_t)l, 0, 0);
#else
  *(float4*)(void*)l = *(const float4*)(const void*)g;
#endif
}

__device__ __forceinline__ void wait_async_all() {
#if HAVE_ASYNC
#if __has_builtin(__builtin_amdgcn_s_wait_asynccnt)
  __builtin_amdgcn_s_wait_asynccnt(0);
#else
  asm volatile("s_wait_asynccnt 0x0" ::: "memory");
#endif
#endif
}

// ---------------------------------------------------------------------------
// Kernel 1: Gaussian scores + softmax.  One block per (i,j,head).
// ---------------------------------------------------------------------------
__global__ __launch_bounds__(256) void gsa_softmax(
    const float* __restrict__ centers,   // [9][2]
    const float* __restrict__ spreads,   // [9]
    float* __restrict__ probs)           // [48*48*9][2304]
{
  const int bid  = blockIdx.x;           // 0..20735
  const int head = bid % NHEADS;
  const int ij   = bid / NHEADS;
  const int i    = ij / HW;
  const int j    = ij % HW;

  const float sp  = spreads[head];
  const float a   = sp * sp;
  const float mu1 = centers[head * 2 + 0];
  const float mu2 = centers[head * 2 + 1];

  const int t = threadIdx.x;
  float sc[9];
  float mx = -1e30f;
#pragma unroll
  for (int it = 0; it < 9; ++it) {
    const int kk = t + 256 * it;
    const int k  = kk / HW;
    const int l  = kk % HW;
    const float dx = (float)(k - i);
    const float dy = (float)(l - j);
    const float s  = a * (mu1 * dx + mu2 * dy - 0.5f * (dx * dx + dy * dy));
    sc[it] = s;
    mx = fmaxf(mx, s);
  }
#pragma unroll
  for (int off = 16; off > 0; off >>= 1) mx = fmaxf(mx, __shfl_xor(mx, off, 32));
  __shared__ float smem[8];
  const int lane = t & 31, wv = t >> 5;
  if (lane == 0) smem[wv] = mx;
  __syncthreads();
  float bm = smem[0];
#pragma unroll
  for (int w = 1; w < 8; ++w) bm = fmaxf(bm, smem[w]);

  float sum = 0.f;
#pragma unroll
  for (int it = 0; it < 9; ++it) { sc[it] = __expf(sc[it] - bm); sum += sc[it]; }
#pragma unroll
  for (int off = 16; off > 0; off >>= 1) sum += __shfl_xor(sum, off, 32);
  __syncthreads();
  if (lane == 0) smem[wv] = sum;
  __syncthreads();
  float bs = 0.f;
#pragma unroll
  for (int w = 0; w < 8; ++w) bs += smem[w];
  const float inv = 1.f / bs;

  float* pr = probs + (size_t)bid * KDIM;
#pragma unroll
  for (int it = 0; it < 9; ++it) pr[t + 256 * it] = sc[it] * inv;
}

// ---------------------------------------------------------------------------
// Pack kernels: fp32 -> bf16 (hi, lo) pairs, N-major ([n][kk]) layouts.
// ---------------------------------------------------------------------------
__global__ __launch_bounds__(256) void gsa_pack_b1(
    const float* __restrict__ hidden,    // [8][256][48][48]  (b,d,l,k)
    __bf16* __restrict__ Bh, __bf16* __restrict__ Bl)  // [n=b*256+d][kk=k*48+l]
{
  const int f  = blockIdx.x * 256 + threadIdx.x;  // coalesced on output
  const int n  = f / KDIM;
  const int kk = f % KDIM;
  const int k  = kk / HW;
  const int l  = kk % HW;
  const float v = hidden[(size_t)n * KDIM + l * HW + k];
  const __bf16 h = (__bf16)v;
  Bh[f] = h;
  Bl[f] = (__bf16)(v - (float)h);
}

__global__ __launch_bounds__(256) void gsa_pack_w(
    const float* __restrict__ W,         // [256][2304] value_w (already [c][kk])
    __bf16* __restrict__ Wh, __bf16* __restrict__ Wl)
{
  const int f = blockIdx.x * 256 + threadIdx.x;   // fully coalesced
  const float v = W[f];
  const __bf16 h = (__bf16)v;
  Wh[f] = h;
  Wl[f] = (__bf16)(v - (float)h);
}

// ---------------------------------------------------------------------------
// WMMA GEMM core.  Block tile 128(M) x 64(N), 8 waves, each 16 x 64.
// A: fp32 row-major (stride KDIM), converted to bf16 hi/lo in-register.
// B: bf16 hi/lo, N-major [n][kk] (row stride KDIM).  K-slab (64 rows x 32 K)
//    double-buffered in LDS via one async b128 copy per thread per array.
// LDS rows padded to 48 bf16 (96B) to reduce ds bank conflicts; the 4 data
// chunks per row sit in the first 64B, keeping 32B v16bf fragment alignment.
// Error-compensated product: Ah*Bh + Ah*Bl + Al*Bh.
// ---------------------------------------------------------------------------
#define GEMM_BODY(EPILOGUE)                                                   \
  __shared__ __align__(32) __bf16 sBh[2][64 * 48];                            \
  __shared__ __align__(32) __bf16 sBl[2][64 * 48];                            \
  const int t    = threadIdx.x;                                               \
  const int lane = t & 31;                                                    \
  const int wv   = t >> 5;                                                    \
  const int m0w  = blockIdx.x * 128 + wv * 16;                                \
  const int nbase = blockIdx.y * 64;                                          \
  /* slab copy assignment: thread -> (row nl_f, 16B chunk part) */            \
  const int nl_f = t >> 2, part = t & 3;                                      \
  const __bf16* gBh = Bh + (size_t)(nbase + nl_f) * KDIM + part * 8;          \
  const __bf16* gBl = Bl + (size_t)(nbase + nl_f) * KDIM + part * 8;          \
  const int ldso = nl_f * 48 + part * 8;                                      \
  /* A fragment addressing */                                                 \
  const int arow   = m0w + (lane & 15);                                       \
  const int kbhalf = 16 * (lane >> 4);                                        \
  const float* Arow = A + (size_t)arow * KDIM;                                \
  v8f acc[4] = {};                                                            \
  copy16_to_lds(gBh, &sBh[0][ldso]);                                          \
  copy16_to_lds(gBl, &sBl[0][ldso]);                                          \
  wait_async_all();                                                           \
  __syncthreads();                                                            \
  int p = 0;                                                                  \
  for (int k0 = 0; k0 < KDIM; k0 += 32) {                                     \
    if (k0 + 32 < KDIM) { /* prefetch next slab into other buffer */          \
      copy16_to_lds(gBh + k0 + 32, &sBh[p ^ 1][ldso]);                        \
      copy16_to_lds(gBl + k0 + 32, &sBl[p ^ 1][ldso]);                        \
    }                                                                         \
    const float4* p0 = (const float4*)(Arow + k0 + kbhalf);                   \
    const float4* p1 = (const float4*)(Arow + k0 + kbhalf + 8);               \
    const float4 a0 = p0[0], a1 = p0[1], a2 = p1[0], a3 = p1[1];              \
    const float af[16] = {a0.x, a0.y, a0.z, a0.w, a1.x, a1.y, a1.z, a1.w,     \
                          a2.x, a2.y, a2.z, a2.w, a3.x, a3.y, a3.z, a3.w};    \
    v16bf ah, al;                                                             \
    _Pragma("unroll")                                                         \
    for (int e = 0; e < 16; ++e) {                                            \
      const __bf16 h = (__bf16)af[e];                                         \
      ah[e] = h;                                                              \
      al[e] = (__bf16)(af[e] - (float)h);                                     \
    }                                                                         \
    const int koff = 16 * (lane >> 4);                                        \
    _Pragma("unroll")                                                         \
    for (int q = 0; q < 4; ++q) {                                             \
      const int nl = q * 16 + (lane & 15);                                    \
      const v16bf bh = *(const v16bf*)(&sBh[p][nl * 48 + koff]);              \
      const v16bf bl = *(const v16bf*)(&sBl[p][nl * 48 + koff]);              \
      acc[q] = __builtin_amdgcn_wmma_f32_16x16x32_bf16(                       \
          false, ah, false, bh, (short)0, acc[q], false, false);              \
      acc[q] = __builtin_amdgcn_wmma_f32_16x16x32_bf16(                       \
          false, ah, false, bl, (short)0, acc[q], false, false);              \
      acc[q] = __builtin_amdgcn_wmma_f32_16x16x32_bf16(                       \
          false, al, false, bh, (short)0, acc[q], false, false);              \
    }                                                                         \
    wait_async_all();                                                         \
    __syncthreads();                                                          \
    p ^= 1;                                                                   \
  }                                                                           \
  EPILOGUE

// GEMM1: vals[b][ij][h][d] = sum_kl probs[ij,h][kl] * hidden[b,d,l,k]
__global__ __launch_bounds__(256) void gsa_gemm1(
    const float* __restrict__ A,                      // probs, M1 x K
    const __bf16* __restrict__ Bh, const __bf16* __restrict__ Bl,  // [N1][K]
    float* __restrict__ vals)                         // [8][2304][9][256]
{
  GEMM_BODY({
    _Pragma("unroll")
    for (int q = 0; q < 4; ++q) {
      const int ng = nbase + q * 16 + (lane & 15);
      const int b  = ng >> 8;
      const int d  = ng & 255;
      _Pragma("unroll")
      for (int r = 0; r < 8; ++r) {
        const int m  = m0w + r + ((lane >> 4) << 3);
        const int ij = m / NHEADS;
        const int h  = m - ij * NHEADS;
        vals[(size_t)b * (KDIM * KDIM) + (size_t)ij * KDIM + h * CH + d] =
            acc[q][r];
      }
    }
  })
}

// GEMM2: out[b][c][j][i] = sum_hd vals[b][ij][hd] * value_w[c][hd] + bias[c]
__global__ __launch_bounds__(256) void gsa_gemm2(
    const float* __restrict__ A,                      // vals, M2 x K
    const __bf16* __restrict__ Bh, const __bf16* __restrict__ Bl,  // [N2][K]
    const float* __restrict__ bias,                   // [256]
    float* __restrict__ out)                          // [8][256][48][48]
{
  GEMM_BODY({
    _Pragma("unroll")
    for (int q = 0; q < 4; ++q) {
      const int c  = nbase + q * 16 + (lane & 15);
      const float bv = bias[c];
      _Pragma("unroll")
      for (int r = 0; r < 8; ++r) {
        const int m  = m0w + r + ((lane >> 4) << 3);
        const int b  = m / KDIM;           // KDIM == 48*48
        const int ij = m - b * KDIM;
        const int i  = ij / HW;
        const int j  = ij % HW;
        out[(size_t)b * (CH * HW * HW) + (size_t)c * (HW * HW) + j * HW + i] =
            acc[q][r] + bv;
      }
    }
  })
}

// ---------------------------------------------------------------------------
extern "C" void kernel_launch(void* const* d_in, const int* in_sizes, int n_in,
                              void* d_out, int out_size, void* d_ws,
                              size_t ws_size, hipStream_t stream) {
  const float* hidden  = (const float*)d_in[0];  // 8*256*48*48
  const float* centers = (const float*)d_in[1];  // 9*2
  const float* spreads = (const float*)d_in[2];  // 9
  const float* value_w = (const float*)d_in[3];  // 256*2304
  const float* value_b = (const float*)d_in[4];  // 256

  float* out   = (float*)d_out;                  // [8][256][48][48]
  float* probs = (float*)d_out + OUT_ELEMS;      // [48][48][9][48][48]

  // workspace layout
  char* ws = (char*)d_ws;
  float*  vals = (float*)ws;                               // 169,869,312 B
  __bf16* B1h  = (__bf16*)(ws + 169869312);                //   9,437,184 B
  __bf16* B1l  = (__bf16*)(ws + 169869312 + 9437184);      //   9,437,184 B
  __bf16* W2h  = (__bf16*)(ws + 169869312 + 2 * 9437184);  //   1,179,648 B
  __bf16* W2l  = (__bf16*)(ws + 169869312 + 2 * 9437184 + 1179648);

  gsa_softmax<<<M1, 256, 0, stream>>>(centers, spreads, probs);
  gsa_pack_b1<<<(N1 * KDIM) / 256, 256, 0, stream>>>(hidden, B1h, B1l);
  gsa_pack_w<<<(N2 * KDIM) / 256, 256, 0, stream>>>(value_w, W2h, W2l);

  gsa_gemm1<<<dim3(M1 / 128, N1 / 64), 256, 0, stream>>>(probs, B1h, B1l, vals);
  gsa_gemm2<<<dim3(M2 / 128, N2 / 64), 256, 0, stream>>>(vals, W2h, W2l,
                                                         value_b, out);
}
